// DGCNN_32444182954696
// MI455X (gfx1250) — matrix operand order repaired and checked by reading, not compiled, not run on hardware
//
#include <hip/hip_runtime.h>

typedef __attribute__((ext_vector_type(16))) _Float16 v16h;
typedef __attribute__((ext_vector_type(8)))  _Float16 v8h;
typedef __attribute__((ext_vector_type(8)))  float    v8f;

static constexpr int    B_   = 4;
static constexpr int    N_   = 4096;
static constexpr int    K_   = 20;
static constexpr size_t COLS = (size_t)B_ * N_ * K_;          // 327680
static constexpr int    CPB  = N_ * K_;                       // cols per batch 81920

// ---------------- workspace layout (bytes) ----------------
static constexpr size_t OFF_IDX   = 0;                        // B*N*K ints   = 1,310,720
static constexpr size_t OFF_STATS = 1310720;                  // 3*128 floats = 1,536
static constexpr size_t OFF_AB1   = OFF_STATS + 1536;         // 4*64 float2  = 2,048
static constexpr size_t OFF_AB2   = OFF_AB1 + 2048;
static constexpr size_t OFF_AB3   = OFF_AB2 + 2048;           // 4*96 float2  = 3,072
static constexpr size_t OFF_W1H   = OFF_AB3 + 3072;           // 64*32 f16
static constexpr size_t OFF_W2H   = OFF_W1H + 4096;           // 64*64 f16
static constexpr size_t OFF_W3H   = OFF_W2H + 8192;           // 96*64 f16
static constexpr size_t OFF_R     = 1344000;                  // 256-aligned big region
static constexpr size_t OFF_FEAT  = OFF_R;                    // 32*COLS f16 = 20,971,520
static constexpr size_t OFF_H1    = OFF_R + 32 * COLS * 2;    // 64*COLS f16 = 41,943,040
static constexpr size_t OFF_H3    = OFF_R;                    // 96*COLS f16 (reuses feat+h1)
static constexpr size_t OFF_H2    = OFF_R + 96 * COLS * 2;    // 64*COLS f16

// ---------------- ds_swizzle xor-butterfly (stays within 16-lane halves) ----------------
template<int XMASK>
__device__ __forceinline__ float swz_add(float v)
{
    // group-of-32 swizzle: and_mask=0x1f, or_mask=0, xor_mask=XMASK
    return v + __int_as_float(
        __builtin_amdgcn_ds_swizzle(__float_as_int(v), (XMASK << 10) | 0x1f));
}
__device__ __forceinline__ float half_reduce(float v)
{
    v = swz_add<1>(v); v = swz_add<2>(v); v = swz_add<4>(v); v = swz_add<8>(v);
    return v;                                                  // total over each 16-lane half
}

// ---------------- prep: f16 weights + zero stats ----------------
__global__ void __launch_bounds__(256)
prep_kernel(const float* __restrict__ W1, const float* __restrict__ W2,
            const float* __restrict__ W3, _Float16* __restrict__ W1h,
            _Float16* __restrict__ W2h, _Float16* __restrict__ W3h,
            float* __restrict__ stats)
{
    int i = blockIdx.x * 256 + threadIdx.x;
    if (i < 384) stats[i] = 0.f;
    if (i < 64 * 32) {
        int m = i >> 5, kk = i & 31;
        W1h[i] = (_Float16)(kk < 19 ? W1[m * 19 + kk] : 0.f);
    }
    if (i < 64 * 64) W2h[i] = (_Float16)W2[i];
    if (i < 96 * 64) W3h[i] = (_Float16)W3[i];
}

// ---------------- kNN: top-20 of pd = 2 x.y - |x|^2 - |y|^2 ----------------
__global__ void __launch_bounds__(256)
knn_kernel(const float* __restrict__ data, int* __restrict__ idx)
{
    __shared__ float sx[N_], sy[N_], sz[N_], sn[N_];          // 64 KB
    const int b = blockIdx.x >> 4;
    const int chunk = blockIdx.x & 15;
    const float* X = data + (size_t)b * 6 * N_;
    for (int m = threadIdx.x; m < N_; m += 256) {
        float x = X[m], y = X[N_ + m], z = X[2 * N_ + m];
        sx[m] = x; sy[m] = y; sz[m] = z; sn[m] = x * x + y * y + z * z;
    }
    __syncthreads();
    const int n = chunk * 256 + threadIdx.x;
    const float qx = sx[n], qy = sy[n], qz = sz[n], qq = sn[n];
    float vals[K_];
    int   vidx[K_];
    for (int t = 0; t < K_; ++t) { vals[t] = -3.4e38f; vidx[t] = 0; }
    for (int m = 0; m < N_; ++m) {
        float d = 2.f * (qx * sx[m] + qy * sy[m] + qz * sz[m]) - qq - sn[m];
        if (d > vals[K_ - 1]) {
            int p = K_ - 1;
            while (p > 0 && vals[p - 1] < d) {                // strict: ties keep lower index first
                vals[p] = vals[p - 1]; vidx[p] = vidx[p - 1]; --p;
            }
            vals[p] = d; vidx[p] = m;
        }
    }
    int* out = idx + ((size_t)b * N_ + n) * K_;
    for (int t = 0; t < K_; ++t) out[t] = vidx[t];
}

// ---------------- geometric feature construction ----------------
__device__ __forceinline__ float angle3(float ax, float ay, float az,
                                        float bx, float by, float bz)
{
    float cx = ay * bz - az * by;
    float cy = az * bx - ax * bz;
    float cz = ax * by - ay * bx;
    float cn = sqrtf(cx * cx + cy * cy + cz * cz);
    return atan2f(cn, ax * bx + ay * by + az * bz);
}

__global__ void __launch_bounds__(256)
feature_kernel(const float* __restrict__ data, const int* __restrict__ idx,
               _Float16* __restrict__ feat)
{
    int i = blockIdx.x * 256 + threadIdx.x;                   // (b,n)
    int b = i >> 12, n = i & (N_ - 1);
    const float* X = data + (size_t)b * 6 * N_;
    float qx = X[n], qy = X[N_ + n], qz = X[2 * N_ + n];
    float rx = X[3 * N_ + n], ry = X[4 * N_ + n], rz = X[5 * N_ + n];
    const int* nb = idx + (size_t)i * K_;
    float gx[K_], gy[K_], gz[K_];
    float cx = 0.f, cy = 0.f, cz = 0.f;
    #pragma unroll 1
    for (int t = 0; t < K_; ++t) {
        int m = nb[t];
        gx[t] = X[m]; gy[t] = X[N_ + m]; gz[t] = X[2 * N_ + m];
        cx += gx[t]; cy += gy[t]; cz += gz[t];
    }
    cx *= 0.05f; cy *= 0.05f; cz *= 0.05f;
    float rcx = cx - qx, rcy = cy - qy, rcz = cz - qz;        // nrnc (same for all k)
    float rcn = sqrtf(rcx * rcx + rcy * rcy + rcz * rcz);
    #pragma unroll 1
    for (int t = 0; t < K_; ++t) {
        int m = nb[t];
        float nx = X[3 * N_ + m], ny = X[4 * N_ + m], nz = X[5 * N_ + m];
        float lx = gx[t] - qx, ly = gy[t] - qy, lz = gz[t] - qz;   // nlxyz
        float ex = gx[t] - cx, ey = gy[t] - cy, ez = gz[t] - cz;   // ncni
        float ln = sqrtf(lx * lx + ly * ly + lz * lz);
        float f[19];
        f[0] = gx[t]; f[1] = gy[t]; f[2] = gz[t];
        f[3] = qx;    f[4] = qy;    f[5] = qz;
        f[6] = lx;    f[7] = ly;    f[8] = lz;
        f[9]  = angle3(rx, ry, rz, lx, ly, lz);
        f[10] = angle3(nx, ny, nz, lx, ly, lz);
        f[11] = angle3(rx, ry, rz, nx, ny, nz);
        f[12] = ln;
        f[13] = rcn;
        f[14] = sqrtf(ex * ex + ey * ey + ez * ez);
        f[15] = ln;                                           // |xc - ngxyz| == |nlxyz|
        f[16] = angle3(rcx, rcy, rcz, lx, ly, lz);            // angle(nrnc, -ninr)
        f[17] = angle3(ex, ey, ez, -rcx, -rcy, -rcz);         // angle(ncni, -nrnc)
        f[18] = angle3(-lx, -ly, -lz, -ex, -ey, -ez);         // angle(ninr, -ncni)
        _Float16* o = feat + ((size_t)i * K_ + t) * 32;
        #pragma unroll
        for (int c2 = 0; c2 < 19; ++c2) o[c2] = (_Float16)f[c2];
        #pragma unroll
        for (int c2 = 19; c2 < 32; ++c2) o[c2] = (_Float16)0.f;
    }
}

// ---------------- fused GEMM (+optional input GN/ReLU) + GN stats ----------------
// Y[col][ch] = sum_k Wh[ch][k] * norm(X[col][k]) ; stats[b][g] += {sum, sumsq}
template<int KSTEPS, int MTILES, int SRCPAD, int DSTPAD, bool NORM>
__global__ void __launch_bounds__(256)
gemm_gn(const _Float16* __restrict__ X, const _Float16* __restrict__ Wh,
        const float2* __restrict__ ab, _Float16* __restrict__ Y,
        float* __restrict__ stats, int colsPerBatch)
{
    constexpr int CIN = KSTEPS * 32;
    constexpr int CPG = MTILES;                                // (MTILES*16)/16 groups
    __shared__ float s_sum[16], s_sq[16];
    if (threadIdx.x < 16) { s_sum[threadIdx.x] = 0.f; s_sq[threadIdx.x] = 0.f; }
    __syncthreads();

    const int wave = threadIdx.x >> 5;
    const int lane = threadIdx.x & 31;
    const int tile = blockIdx.x * 8 + wave;
    const int col0 = tile * 16;
    const int colB = col0 + (lane & 15);
    const int b    = col0 / colsPerBatch;
    const int hi   = lane >> 4;

    // B operand: lane holds column colB, K = s*32 + hi*16 + j (contiguous 16 halves)
    v16h Bm[KSTEPS];
    #pragma unroll
    for (int s = 0; s < KSTEPS; ++s) {
        const int kbase = s * 32 + hi * 16;
        v16h v = *(const v16h*)(X + (size_t)colB * SRCPAD + kbase);
        if (NORM) {
            #pragma unroll
            for (int j = 0; j < 16; ++j) {
                float2 t = ab[b * CIN + kbase + j];
                float f = (float)v[j] * t.x + t.y;
                v[j] = (_Float16)fmaxf(f, 0.f);
            }
        }
        Bm[s] = v;
    }

    const int akb = hi * 16;                                   // A-operand K sub-base
    #pragma unroll
    for (int mt = 0; mt < MTILES; ++mt) {
        v8f c = {};
        #pragma unroll
        for (int s = 0; s < KSTEPS; ++s) {
            const _Float16* w = Wh + (size_t)((lane & 15) + mt * 16) * CIN + s * 32 + akb;
            v8h lo  = *(const v8h*)(w);
            v8h hiv = *(const v8h*)(w + 8);
            v16h a;
            #pragma unroll
            for (int j = 0; j < 8; ++j) { a[j] = lo[j]; a[j + 8] = hiv[j]; }
            c = __builtin_amdgcn_wmma_f32_16x16x32_f16(false, a, false, Bm[s],
                                                       (short)0, c, false, false);
        }
        const int ch0 = mt * 16 + hi * 8;                      // 8 consecutive channels
        // ---- store f16 result + lane-local stats (always exactly 2 groups) ----
        const int gbase = ch0 / CPG;
        float s0 = 0.f, s1 = 0.f, q0 = 0.f, q1 = 0.f;
        v8h outv;
        #pragma unroll
        for (int r = 0; r < 8; ++r) {
            float f = c[r];
            outv[r] = (_Float16)f;
            if ((ch0 + r) / CPG == gbase) { s0 += f; q0 += f * f; }  // const fold CPG==4
            else                          { s1 += f; q1 += f * f; }
        }
        *(v8h*)(Y + (size_t)colB * DSTPAD + ch0) = outv;
        // ---- ds_swizzle butterfly over the 16-lane half (identical channel sets) ----
        s0 = half_reduce(s0); q0 = half_reduce(q0);
        s1 = half_reduce(s1); q1 = half_reduce(q1);
        if ((lane & 15) == 0) {                                // lanes 0 and 16 flush
            atomicAdd(&s_sum[gbase], s0);
            atomicAdd(&s_sq [gbase], q0);
            atomicAdd(&s_sum[gbase + 1], s1);
            atomicAdd(&s_sq [gbase + 1], q1);
        }
    }
    __syncthreads();
    if (threadIdx.x < 16) {
        atomicAdd(&stats[(b * 16 + threadIdx.x) * 2 + 0], s_sum[threadIdx.x]);
        atomicAdd(&stats[(b * 16 + threadIdx.x) * 2 + 1], s_sq [threadIdx.x]);
    }
}

// ---------------- per-(batch,channel) GN affine coefficients ----------------
template<int C>
__global__ void __launch_bounds__(256)
finalize_kernel(const float* __restrict__ stats, const float* __restrict__ gamma,
                const float* __restrict__ beta, float2* __restrict__ ab, float invCount)
{
    int i = blockIdx.x * blockDim.x + threadIdx.x;
    if (i >= 4 * C) return;
    int b = i / C, ch = i % C;
    int g = ch / (C / 16);
    float sum = stats[(b * 16 + g) * 2 + 0];
    float sq  = stats[(b * 16 + g) * 2 + 1];
    float mean = sum * invCount;
    float var  = sq * invCount - mean * mean;
    float inv  = rsqrtf(var + 1e-5f);
    float a = gamma[ch] * inv;
    ab[i] = make_float2(a, beta[ch] - mean * a);
}

// ---------------- fused GN3 + ReLU + max over k (8 channels / thread) ----------------
__global__ void __launch_bounds__(256)
maxk_kernel(const _Float16* __restrict__ H3, const float2* __restrict__ ab,
            float* __restrict__ out)
{
    int i  = blockIdx.x * 256 + threadIdx.x;                   // (b, c8, n): 4*12*4096
    int n  = i & (N_ - 1);
    int c8 = (i >> 12) % 12;
    int b  = i / (12 * N_);
    const int c0 = c8 * 8;
    float2 t[8];
    #pragma unroll
    for (int r = 0; r < 8; ++r) t[r] = ab[b * 96 + c0 + r];
    size_t colbase = ((size_t)b * N_ + n) * K_;
    float m[8];
    #pragma unroll
    for (int r = 0; r < 8; ++r) m[r] = 0.f;                    // relu output >= 0
    #pragma unroll 1
    for (int kk = 0; kk < K_; ++kk) {
        v8h h = *(const v8h*)(H3 + (colbase + kk) * 96 + c0);  // 16B aligned
        #pragma unroll
        for (int r = 0; r < 8; ++r) {
            float v = (float)h[r] * t[r].x + t[r].y;
            m[r] = fmaxf(m[r], fmaxf(v, 0.f));
        }
    }
    float* o = out + ((size_t)b * 96 + c0) * N_ + n;
    #pragma unroll
    for (int r = 0; r < 8; ++r) o[(size_t)r * N_] = m[r];
}

// ---------------- host launcher ----------------
extern "C" void kernel_launch(void* const* d_in, const int* in_sizes, int n_in,
                              void* d_out, int out_size, void* d_ws, size_t ws_size,
                              hipStream_t stream)
{
    const float* data = (const float*)d_in[0];
    const float* W1 = (const float*)d_in[1];
    const float* g1 = (const float*)d_in[2];
    const float* b1 = (const float*)d_in[3];
    const float* W2 = (const float*)d_in[4];
    const float* g2 = (const float*)d_in[5];
    const float* b2 = (const float*)d_in[6];
    const float* W3 = (const float*)d_in[7];
    const float* g3 = (const float*)d_in[8];
    const float* b3 = (const float*)d_in[9];
    // d_in[10] is k == 20, baked into K_.

    char* ws = (char*)d_ws;
    int*      idx   = (int*)     (ws + OFF_IDX);
    float*    stats = (float*)   (ws + OFF_STATS);
    float2*   ab1   = (float2*)  (ws + OFF_AB1);
    float2*   ab2   = (float2*)  (ws + OFF_AB2);
    float2*   ab3   = (float2*)  (ws + OFF_AB3);
    _Float16* W1h   = (_Float16*)(ws + OFF_W1H);
    _Float16* W2h   = (_Float16*)(ws + OFF_W2H);
    _Float16* W3h   = (_Float16*)(ws + OFF_W3H);
    _Float16* feat  = (_Float16*)(ws + OFF_FEAT);
    _Float16* h1    = (_Float16*)(ws + OFF_H1);
    _Float16* h2    = (_Float16*)(ws + OFF_H2);
    _Float16* h3    = (_Float16*)(ws + OFF_H3);

    const int tiles = (int)(COLS / 16);                        // 20480
    const int gwgs  = tiles / 8;                               // 2560

    prep_kernel<<<24, 256, 0, stream>>>(W1, W2, W3, W1h, W2h, W3h, stats);
    knn_kernel<<<64, 256, 0, stream>>>(data, idx);
    feature_kernel<<<64, 256, 0, stream>>>(data, idx, feat);

    gemm_gn<1, 4, 32, 64, false><<<gwgs, 256, 0, stream>>>(feat, W1h, nullptr, h1,
                                                           stats + 0, CPB);
    finalize_kernel<64><<<1, 256, 0, stream>>>(stats + 0, g1, b1, ab1,
                                               1.f / (4.f * N_ * K_));
    gemm_gn<2, 4, 64, 64, true><<<gwgs, 256, 0, stream>>>(h1, W2h, ab1, h2,
                                                          stats + 128, CPB);
    finalize_kernel<64><<<1, 256, 0, stream>>>(stats + 128, g2, b2, ab2,
                                               1.f / (4.f * N_ * K_));
    gemm_gn<2, 6, 64, 96, true><<<gwgs, 256, 0, stream>>>(h2, W3h, ab2, h3,
                                                          stats + 256, CPB);
    finalize_kernel<96><<<2, 256, 0, stream>>>(stats + 256, g3, b3, ab3,
                                               1.f / (6.f * N_ * K_));
    maxk_kernel<<<(4 * 12 * N_) / 256, 256, 0, stream>>>(h3, ab3, (float*)d_out);
}
